// VectorizedExpertMLP_28312424415696
// MI455X (gfx1250) — matrix operand order repaired and checked by reading, not compiled, not run on hardware
//
#include <hip/hip_runtime.h>

// MoE top-2 expert MLP, fused grouped-GEMM with bf16 WMMA + TDM on gfx1250.
//   x:[T,D] f32, rw:[T,K] f32, se:[T,K] i32, w1:[E,D,H] f32, w2:[E,H,D] f32
//   out[t] = sum_k rw[t,k] * silu(x[t] @ w1[e]) @ w2[e],  e = se[t,k]

typedef __attribute__((ext_vector_type(16))) __bf16       v16bf;
typedef __attribute__((ext_vector_type(8)))  __bf16       v8bf;
typedef __attribute__((ext_vector_type(8)))  float        v8f;
typedef __attribute__((ext_vector_type(8)))  unsigned int v8u;
typedef __attribute__((ext_vector_type(4)))  unsigned int v4u;
typedef __attribute__((ext_vector_type(8)))  int          v8i;
typedef __attribute__((ext_vector_type(4)))  int          v4i;

#define T_TOK  2048
#define DMODEL 1024
#define HDIM   4096
#define NEXP   8
#define TOPK   2
#define CAP    (T_TOK * TOPK)   // 4096 token-slots max per expert
#define MT     32               // tokens per workgroup tile
#define HC     64               // H chunk per fused iteration
#define KB     128              // K rows per TDM weight slab
#define NSLAB  (DMODEL / KB)    // 8 slabs per chunk
#define XS_STRIDE 1032          // bf16 halves per Xs row (bank-conflict-free)
#define HS_STRIDE 72            // bf16 halves per Hs row
#define WF_STRIDE 67            // f32 per W-slab row (64 + 3 pad via TDM padding)
#define TILES_PER_E (CAP / MT)  // 128

// ---------------- helpers ----------------

__device__ __forceinline__ __bf16 f2bf(float f) {
  unsigned int u = __builtin_bit_cast(unsigned int, f);
  return __builtin_bit_cast(__bf16, (unsigned short)(u >> 16));
}

// Pack 16 f32 bit patterns -> 16 bf16 (truncate) using v_perm_b32 pairs.
__device__ __forceinline__ v16bf pack16(const unsigned int* t) {
  v8u r;
#pragma unroll
  for (int h = 0; h < 8; ++h)   // result = {hi16(t[2h+1]), hi16(t[2h])}
    r[h] = __builtin_amdgcn_perm(t[2 * h + 1], t[2 * h], 0x07060302);
  return __builtin_bit_cast(v16bf, r);
}

// A fragment 16x32 bf16 from LDS (row-major, 16B-aligned rows).
// Lanes 0-15 (M=lane): K=0..7,16..23; lanes 16-31: K=8..15,24..31.
__device__ __forceinline__ v16bf load_a_lds(const __bf16* base, int stride,
                                            int k0, int lane) {
  const int m  = lane & 15;
  const int kb = (lane >> 4) << 3;          // 0 or 8
  const __bf16* p = base + m * stride + k0 + kb;
  v8bf lo = *(const v8bf*)(p);              // ds_load_b128
  v8bf hi = *(const v8bf*)(p + 16);
  return __builtin_shufflevector(lo, hi, 0,1,2,3,4,5,6,7,8,9,10,11,12,13,14,15);
}

// B fragment 32x16 from f32 matrix (row-major [K][N], leading dim ld) -> bf16.
// Lanes 0-15: N=lane, K=0..7,16..23; lanes 16-31: N=lane-16, K=8..15,24..31.
// Loads go to a flat temp array first so they clause with ONE wait, then pack.
__device__ __forceinline__ v16bf load_b_f32(const float* __restrict__ w, int ld,
                                            int k0, int n0, int lane) {
  const int n  = lane & 15;
  const int kb = (lane >> 4) << 3;          // 0 or 8
  const float* p = w + (size_t)(k0 + kb) * ld + (n0 + n);
  unsigned int t[16];
#pragma unroll
  for (int h = 0; h < 16; ++h)
    t[h] = __builtin_bit_cast(unsigned int, p[(size_t)(h < 8 ? h : h + 8) * ld]);
  return pack16(t);
}

__device__ __forceinline__ v8f wmma_bf16(v16bf a, v16bf b, v8f c) {
  return __builtin_amdgcn_wmma_f32_16x16x32_bf16(false, a, false, b,
                                                 (short)0, c, false, false);
}

// TDM: DMA a [KB x 64] f32 slab (row stride HDIM elements) from global into
// LDS at byte address lds_off, padding each 64-DWORD row with 3 DWORDs so LDS
// rows land at stride WF_STRIDE=67 floats (conflict-free strided B reads).
__device__ __forceinline__ void tdm_load_w1_slab(const float* gsrc,
                                                 unsigned lds_off) {
  unsigned long long ga = (unsigned long long)(const void*)gsrc;
  v4u g0;
  g0[0] = 1u;                                            // count=1, user mode
  g0[1] = lds_off;                                       // LDS byte address
  g0[2] = (unsigned)(ga & 0xFFFFFFFFu);                  // global_addr[31:0]
  g0[3] = (unsigned)((ga >> 32) & 0x1FFFFFFu) | (2u << 30); // [56:32] | type=2
  v8i g1;
  g1[0] = (int)((2u << 16)      // data_size = 4B
              | (1u << 20)      // pad_enable
              | (5u << 22)      // pad_interval = 64 DWORDs
              | (2u << 25));    // pad_amount = 3 DWORDs
  g1[1] = (int)(((unsigned)HDIM & 0xFFFFu) << 16);       // tensor_dim0 lo
  g1[2] = (int)(((unsigned)HDIM >> 16) |
                (((unsigned)DMODEL & 0xFFFFu) << 16));   // dim0 hi | dim1 lo
  g1[3] = (int)(((unsigned)DMODEL >> 16) | (64u << 16)); // dim1 hi | tile_dim0
  g1[4] = (int)(KB);                                     // tile_dim1 (rows)
  g1[5] = (int)(HDIM);                                   // tensor_dim0_stride
  g1[6] = 0;
  g1[7] = 0;
  v4i z4 = {0, 0, 0, 0};                                 // 2D: groups 2/3 unused
  v8i z8 = {0, 0, 0, 0, 0, 0, 0, 0};                     // unused trailing group
  __builtin_amdgcn_tensor_load_to_lds(g0, g1, z4, z4, z8, 0);
}

// ---------------- pass 1: routing scatter ----------------

__global__ void moe_route_kernel(const int* __restrict__ se,
                                 const float* __restrict__ rw,
                                 int* __restrict__ counts,
                                 int* __restrict__ tok,
                                 float* __restrict__ wts) {
  int i = blockIdx.x * blockDim.x + threadIdx.x;
  if (i >= T_TOK * TOPK) return;
  int e = se[i];
  int t = i / TOPK;
  int pos = atomicAdd(&counts[e], 1);
  tok[e * CAP + pos] = t;
  wts[e * CAP + pos] = rw[i];
}

// ---------------- pass 2: fused expert MLP ----------------

__global__ __launch_bounds__(256) void moe_mlp_kernel(
    const float* __restrict__ x,  const float* __restrict__ w1,
    const float* __restrict__ w2, const int* __restrict__ counts,
    const int* __restrict__ tok,  const float* __restrict__ wts,
    float* __restrict__ out) {
  __shared__ __align__(16) __bf16 Xs[MT * XS_STRIDE];     // 32x1024 bf16 (66KB)
  __shared__ __align__(16) __bf16 Hs[MT * HS_STRIDE];     // 32x64 hidden chunk
  __shared__ __align__(16) float  Wf[2][KB * WF_STRIDE];  // TDM slab dbl-buf
  __shared__ int   stok[MT];
  __shared__ float swt[MT];

  const int e    = blockIdx.x / TILES_PER_E;
  const int tile = blockIdx.x % TILES_PER_E;
  const int cnt  = counts[e];
  if (tile * MT >= cnt) return;

  const int tid  = threadIdx.x;
  const int lane = tid & 31;
  const int wave = tid >> 5;               // 8 waves

  if (tid < MT) {
    int idx = tile * MT + tid;
    int tv  = (idx < cnt) ? tok[e * CAP + idx] : -1;
    stok[tid] = tv;
    swt[tid]  = (idx < cnt) ? wts[e * CAP + idx] : 0.0f;
  }
  __syncthreads();

  // Stage X tile into LDS as bf16: 8 threads per row, 128 cols each.
  {
    const int r  = tid >> 3;               // 0..31
    const int c0 = (tid & 7) * 128;
    const int tv = stok[r];
    __bf16* dst = &Xs[r * XS_STRIDE + c0];
    if (tv >= 0) {
      const float4* xr = (const float4*)(x + (size_t)tv * DMODEL + c0);
#pragma unroll 8
      for (int i = 0; i < 32; ++i) {
        float4 v = xr[i];
        unsigned u0 = __builtin_bit_cast(unsigned, v.x);
        unsigned u1 = __builtin_bit_cast(unsigned, v.y);
        unsigned u2 = __builtin_bit_cast(unsigned, v.z);
        unsigned u3 = __builtin_bit_cast(unsigned, v.w);
        uint2 pk;
        pk.x = __builtin_amdgcn_perm(u1, u0, 0x07060302);
        pk.y = __builtin_amdgcn_perm(u3, u2, 0x07060302);
        *(uint2*)(dst + i * 4) = pk;       // 4 bf16 per 8B store
      }
    } else {
      uint2 z; z.x = 0; z.y = 0;
#pragma unroll 8
      for (int i = 0; i < 32; ++i) *(uint2*)(dst + i * 4) = z;
    }
  }
  __syncthreads();

  const int mg = wave >> 2;                // token group: rows mg*16 .. +15
  const int ng = wave & 3;                 // stage1: H-tile; stage2: 256-col grp

  const float* W1e = w1 + (size_t)e * DMODEL * HDIM;   // [D][H]
  const float* W2e = w2 + (size_t)e * HDIM * DMODEL;   // [H][D]

  // Persistent output accumulators: [16 tokens x 256 D-cols] = 16 frags.
  v8f cfr[16];
#pragma unroll
  for (int i = 0; i < 16; ++i)
#pragma unroll
    for (int j = 0; j < 8; ++j) cfr[i][j] = 0.0f;

  const int half = lane >> 4;              // C/D frag: row = j + 8*half
  const int nn   = lane & 15;              // C/D frag: col = lane & 15

  // Low 32 bits of a flat LDS pointer == wave-relative LDS byte address.
  const unsigned wf_off[2] = {(unsigned)(size_t)(const void*)&Wf[0][0],
                              (unsigned)(size_t)(const void*)&Wf[1][0]};

  // Prime the TDM pipeline: slab 0 of chunk 0 (rows 0..KB-1, cols 0..63).
  if (wave == 0) tdm_load_w1_slab(W1e, wf_off[0]);

  for (int hc = 0; hc < HDIM; hc += HC) {
    // ---- stage 1: Hs = silu(X @ W1[:, hc:hc+64]), TDM-fed, dbl-buffered ----
    v8f acc;
#pragma unroll
    for (int j = 0; j < 8; ++j) acc[j] = 0.0f;

    for (int s = 0; s < NSLAB; ++s) {
      if (wave == 0) {
        if (s + 1 < NSLAB) {               // next slab of this chunk
          tdm_load_w1_slab(W1e + hc + (size_t)(s + 1) * KB * HDIM,
                           wf_off[(s + 1) & 1]);
          __builtin_amdgcn_s_wait_tensorcnt(1);
        } else if (hc + HC < HDIM) {       // first slab of next chunk:
          tdm_load_w1_slab(W1e + (hc + HC), wf_off[0]);  // overlaps stage 2
          __builtin_amdgcn_s_wait_tensorcnt(1);
        } else {
          __builtin_amdgcn_s_wait_tensorcnt(0);
        }
      }
      __syncthreads();                     // slab s resident in Wf[s&1]
      const float* wf = &Wf[s & 1][0];
#pragma unroll
      for (int kk = 0; kk < KB; kk += 32) {
        v16bf a = load_a_lds(&Xs[mg * 16 * XS_STRIDE], XS_STRIDE,
                             s * KB + kk, lane);
        v16bf b = load_b_f32(wf, WF_STRIDE, kk, ng * 16, lane);
        acc = wmma_bf16(a, b, acc);
      }
      __syncthreads();                     // all waves done with Wf[s&1]
    }

#pragma unroll
    for (int j = 0; j < 8; ++j) {
      float v = acc[j];
      float sil = v / (1.0f + __expf(-v)); // silu
      Hs[(mg * 16 + j + 8 * half) * HS_STRIDE + ng * 16 + nn] = f2bf(sil);
    }
    __syncthreads();

    // ---- stage 2: cfr += Hs[mg rows] @ W2[hc:hc+64, ng*256:+256] ----
#pragma unroll
    for (int k2 = 0; k2 < HC; k2 += 32) {
      if (hc + k2 + 32 < HDIM)
        __builtin_prefetch(W2e + (size_t)(hc + k2 + 32) * DMODEL + ng * 256,
                           0, 1);
      v16bf a2 = load_a_lds(&Hs[mg * 16 * HS_STRIDE], HS_STRIDE, k2, lane);
#pragma unroll
      for (int nt = 0; nt < 16; ++nt) {
        v16bf b2 = load_b_f32(W2e, DMODEL, hc + k2, ng * 256 + nt * 16, lane);
        cfr[nt] = wmma_bf16(a2, b2, cfr[nt]);
      }
    }
    __syncthreads();
  }

  // ---- writeout: scale by routing weight, accumulate into out ----
#pragma unroll
  for (int nt = 0; nt < 16; ++nt) {
    const int col = ng * 256 + nt * 16 + nn;
#pragma unroll
    for (int j = 0; j < 8; ++j) {
      const int m  = mg * 16 + j + 8 * half;
      const int tv = stok[m];
      if (tv >= 0)
        atomicAdd(out + (size_t)tv * DMODEL + col, cfr[nt][j] * swt[m]);
    }
  }
}

// ---------------- launch ----------------

extern "C" void kernel_launch(void* const* d_in, const int* in_sizes, int n_in,
                              void* d_out, int out_size, void* d_ws, size_t ws_size,
                              hipStream_t stream) {
  const float* x  = (const float*)d_in[0];
  const float* rw = (const float*)d_in[1];
  const int*   se = (const int*)d_in[2];
  const float* w1 = (const float*)d_in[3];
  const float* w2 = (const float*)d_in[4];
  float* out = (float*)d_out;

  // workspace: [counts: pad 1KB] [tok: E*CAP ints] [wts: E*CAP floats]
  int*   counts = (int*)d_ws;
  int*   tok    = (int*)((char*)d_ws + 1024);
  float* wts    = (float*)((char*)d_ws + 1024 + (size_t)NEXP * CAP * sizeof(int));

  (void)hipMemsetAsync(d_out, 0, (size_t)out_size * sizeof(float), stream);
  (void)hipMemsetAsync(counts, 0, 1024, stream);

  moe_route_kernel<<<(T_TOK * TOPK + 255) / 256, 256, 0, stream>>>(
      se, rw, counts, tok, wts);

  moe_mlp_kernel<<<NEXP * TILES_PER_E, 256, 0, stream>>>(
      x, w1, w2, counts, tok, wts, out);
}
